// MultiHeadAttention_83442624626716
// MI455X (gfx1250) — compile-verified, matrix operand order checked by reading, not run on hardware
//
#include <hip/hip_runtime.h>
#include <hip/hip_bf16.h>
#include <math.h>

#define B_ 4
#define T_ 2048
#define C_ 1024
#define H_ 16
#define HD_ 64
#define SCALE_ 0.125f

typedef __attribute__((ext_vector_type(16))) __bf16 v16bf;
typedef __attribute__((ext_vector_type(8)))  float  v8f;
typedef __attribute__((ext_vector_type(8)))  unsigned short v8u16;

union FragU { v16bf v; v8u16 h[2]; };

__device__ __forceinline__ unsigned short f2bf(float f) {
  unsigned int u = __float_as_uint(f);
  u += 0x7FFFu + ((u >> 16) & 1u);          // round-to-nearest-even
  return (unsigned short)(u >> 16);
}

__device__ __forceinline__ v8f v8f_zero() {
  v8f v;
#pragma unroll
  for (int i = 0; i < 8; ++i) v[i] = 0.0f;
  return v;
}

// Load one 16x32 bf16 fragment (A-style lane layout) from a row-major
// [rows][ld] bf16 matrix. Two 16B loads per lane (global or LDS).
__device__ __forceinline__ v16bf load_frag(const unsigned short* p0, int row,
                                           int ld, int kb, int lane) {
  const int l = lane & 15, hi = lane >> 4;
  const unsigned short* p = p0 + (size_t)(row + l) * (size_t)ld + kb + hi * 8;
  FragU f;
  f.h[0] = *(const v8u16*)(p);
  f.h[1] = *(const v8u16*)(p + 16);
  return f.v;
}

__device__ __forceinline__ v8f wmma_bf16(v16bf a, v16bf b, v8f c) {
  return __builtin_amdgcn_wmma_f32_16x16x32_bf16(
      /*neg_a=*/false, a, /*neg_b=*/false, b,
      /*c_mod=*/(short)0, c, /*reuse_a=*/false, /*reuse_b=*/false);
}

// 32-bit LDS byte offset of a __shared__ object (generic ptr low bits = LDS
// aperture offset on gfx1250)
__device__ __forceinline__ unsigned lds_addr32(const void* p) {
  return (unsigned)(unsigned long long)(uintptr_t)p;
}

// Async global -> LDS copy, 16 bytes per enabled lane; tracked by ASYNCcnt.
__device__ __forceinline__ void async_g2l_b128(unsigned lds_off,
                                               const unsigned short* gaddr) {
  asm volatile("global_load_async_to_lds_b128 %0, %1, off"
               :: "v"(lds_off), "v"(gaddr) : "memory");
}

// ---------------------------------------------------------------- converts
__global__ void k_cvt(const float* __restrict__ s, unsigned short* __restrict__ d, int n) {
  int i = blockIdx.x * blockDim.x + threadIdx.x;
  const int stride = gridDim.x * blockDim.x;
  for (; i < n; i += stride) d[i] = f2bf(s[i]);
}

// src [K][N] row-major fp32 -> dst [N][K] bf16 (N-major so B fragments load
// with the same contiguous pattern as A fragments)
__global__ void k_cvt_t(const float* __restrict__ s, unsigned short* __restrict__ d,
                        int K, int N) {
  int i = blockIdx.x * blockDim.x + threadIdx.x;
  const int total = K * N;
  const int stride = gridDim.x * blockDim.x;
  for (; i < total; i += stride) {
    int n = i / K, k = i - n * K;
    d[i] = f2bf(s[(size_t)k * N + n]);
  }
}

// ------------------------------------------------------------- QKV GEMM
// [8192,1024] x [1024,3072].  256 threads = 8 waves arranged 2(M)x4(N),
// 32x64 per wave -> block tile 64x256.  8 WMMAs per 6 fragment loads.
__global__ __launch_bounds__(256) void k_qkv_gemm(
    const unsigned short* __restrict__ xb,   // [8192][1024] bf16
    const unsigned short* __restrict__ wt,   // [3072][1024] bf16 (N-major)
    const float* __restrict__ bias,          // [3072]
    unsigned short* __restrict__ qws,
    unsigned short* __restrict__ kws,
    unsigned short* __restrict__ vtws) {
  const int lane = threadIdx.x & 31;
  const int wave = threadIdx.x >> 5;
  const int wm = wave >> 2, wn = wave & 3;
  const int rm = blockIdx.y * 64 + wm * 32;
  const int cn = blockIdx.x * 256 + wn * 64;
  v8f acc[2][4];
#pragma unroll
  for (int i = 0; i < 2; ++i)
#pragma unroll
    for (int j = 0; j < 4; ++j) acc[i][j] = v8f_zero();

#pragma unroll 2
  for (int k0 = 0; k0 < C_; k0 += 32) {
    v16bf fa[2], fb[4];
#pragma unroll
    for (int i = 0; i < 2; ++i) fa[i] = load_frag(xb, rm + i * 16, C_, k0, lane);
#pragma unroll
    for (int j = 0; j < 4; ++j) fb[j] = load_frag(wt, cn + j * 16, C_, k0, lane);
    if (k0 + 64 < C_) {  // hide L2 latency behind WMMA issue
      __builtin_prefetch(xb + (size_t)(rm + (lane & 15)) * C_ + k0 + 64, 0, 1);
      __builtin_prefetch(wt + (size_t)(cn + (lane & 15)) * C_ + k0 + 64, 0, 1);
    }
#pragma unroll
    for (int i = 0; i < 2; ++i)
#pragma unroll
      for (int j = 0; j < 4; ++j) acc[i][j] = wmma_bf16(fa[i], fb[j], acc[i][j]);
  }

  const int l15 = lane & 15, hs = lane >> 4;
#pragma unroll
  for (int i = 0; i < 2; ++i) {
#pragma unroll
    for (int j = 0; j < 4; ++j) {
#pragma unroll
      for (int r = 0; r < 8; ++r) {
        const int m = rm + i * 16 + hs * 8 + r;
        const int n = cn + j * 16 + l15;
        const float val = acc[i][j][r] + bias[n];
        const int sel = n >> 10, rem = n & 1023;
        const int h = rem >> 6, d = rem & 63;
        const int b = m >> 11, t = m & (T_ - 1);
        const size_t bh = (size_t)(b * H_ + h);
        const unsigned short bv = f2bf(val);
        if (sel == 0)      qws[(bh * T_ + t) * HD_ + d] = bv;
        else if (sel == 1) kws[(bh * T_ + t) * HD_ + d] = bv;
        else               vtws[(bh * HD_ + d) * T_ + t] = bv;
      }
    }
  }
}

// ---------------------------------------------------------- flash attention
// grid: (T/128, B*H); block = 128 = 4 waves; each wave owns 32 query rows
// (2 strips of 16).  K/V tiles for each 32-key step are staged ONCE per
// block into LDS with async global->LDS copies (ASYNCcnt), then all four
// waves consume them via ds_load fragments: 4x less K/V global traffic and
// 16 WMMAs per tile per wave.  Waves loop over the block's max causal range
// (uniform trip count for the barriers) and skip compute (wave-uniform
// branch, EXEC stays all-ones) for tiles beyond their own causal range.
__global__ __launch_bounds__(128) void k_attn(
    const unsigned short* __restrict__ qws,
    const unsigned short* __restrict__ kws,
    const unsigned short* __restrict__ vtws,
    const int* __restrict__ padmask,   // [B][T]
    unsigned short* __restrict__ attn) // [B][T][C] bf16
{
  __shared__ unsigned short ldsK[32 * 64];     // K tile, rows = key, ld = 64
  __shared__ unsigned short ldsV[64 * 32];     // V^T tile, rows = hd, ld = 32
  __shared__ unsigned short pbuf[4][32 * 32];  // per-wave P transpose buffer
  const int tid = threadIdx.x;
  const int lane = tid & 31;
  const int wave = tid >> 5;
  const int l15 = lane & 15, hs = lane >> 4;
  const int bh = blockIdx.y;
  const int b = bh >> 4;
  const int h = bh & 15;
  const int qsB = blockIdx.x * 128;
  const int qs = qsB + wave * 32;

  const unsigned short* qp = qws + (size_t)bh * T_ * HD_;
  const unsigned short* kp = kws + (size_t)bh * T_ * HD_;
  const unsigned short* vp = vtws + (size_t)bh * HD_ * T_;
  const int* pp = padmask + b * T_;
  unsigned short* pb = &pbuf[wave][0];
  const unsigned kA = lds_addr32(ldsK);
  const unsigned vA = lds_addr32(ldsV);

  v16bf aq[2][2];
#pragma unroll
  for (int s = 0; s < 2; ++s) {
    aq[s][0] = load_frag(qp, qs + s * 16, HD_, 0, lane);
    aq[s][1] = load_frag(qp, qs + s * 16, HD_, 32, lane);
  }

  v8f o[2][4];
  float mrow[2][8], lrow[2][8];
#pragma unroll
  for (int s = 0; s < 2; ++s) {
#pragma unroll
    for (int t = 0; t < 4; ++t) o[s][t] = v8f_zero();
#pragma unroll
    for (int r = 0; r < 8; ++r) { mrow[s][r] = -1e30f; lrow[s][r] = 0.0f; }
  }

  for (int kb = 0; kb <= qsB + 127; kb += 32) {
    // ---- cooperative async staging: K (32x64) and V^T (64x32), 4KB each.
    // 256 16B chunks per tile, 128 threads -> 2 chunks each per tile.
#pragma unroll
    for (int c = 0; c < 2; ++c) {
      const int ck = tid + c * 128;
      const int krow = ck >> 3, kcol = ck & 7;        // 8 chunks per 64-el row
      async_g2l_b128(kA + ck * 16, kp + (size_t)(kb + krow) * HD_ + kcol * 8);
      const int vrow = ck >> 2, vcol = ck & 3;        // 4 chunks per 32-el row
      async_g2l_b128(vA + ck * 16, vp + (size_t)vrow * T_ + kb + vcol * 8);
    }
    asm volatile("s_wait_asynccnt 0" ::: "memory");
    __syncthreads();

    if (kb <= qs + 31) {                              // wave-uniform branch
      // S = Q K^T, keys [kb,kb+32): K-frags from LDS, reused by both strips
      v16bf bk00 = load_frag(ldsK, 0,  HD_, 0,  lane);
      v16bf bk01 = load_frag(ldsK, 0,  HD_, 32, lane);
      v16bf bk10 = load_frag(ldsK, 16, HD_, 0,  lane);
      v16bf bk11 = load_frag(ldsK, 16, HD_, 32, lane);
      v8f st[2][2];
#pragma unroll
      for (int s = 0; s < 2; ++s) {
        st[s][0] = v8f_zero();
        st[s][1] = v8f_zero();
        st[s][0] = wmma_bf16(aq[s][0], bk00, st[s][0]);
        st[s][0] = wmma_bf16(aq[s][1], bk01, st[s][0]);
        st[s][1] = wmma_bf16(aq[s][0], bk10, st[s][1]);
        st[s][1] = wmma_bf16(aq[s][1], bk11, st[s][1]);
      }

      const int k0c = kb + l15;
      const int k1c = kb + 16 + l15;
      const int pad0 = pp[k0c];
      const int pad1 = pp[k1c];
#pragma unroll
      for (int s = 0; s < 2; ++s) {
#pragma unroll
        for (int r = 0; r < 8; ++r) {
          const int q = qs + s * 16 + hs * 8 + r;
          float v0 = st[s][0][r] * SCALE_;
          float v1 = st[s][1][r] * SCALE_;
          const bool keep0 = (k0c <= q) && (pad0 != 0);
          const bool keep1 = (k1c <= q) && (pad1 != 0);
          v0 = keep0 ? v0 : -1e30f;
          v1 = keep1 ? v1 : -1e30f;
          // row max across the 16 lanes of this half (xor<=8 stays in-half)
          float tmax = fmaxf(v0, v1);
          tmax = fmaxf(tmax, __shfl_xor(tmax, 1, 32));
          tmax = fmaxf(tmax, __shfl_xor(tmax, 2, 32));
          tmax = fmaxf(tmax, __shfl_xor(tmax, 4, 32));
          tmax = fmaxf(tmax, __shfl_xor(tmax, 8, 32));
          const float mnew = fmaxf(mrow[s][r], tmax);
          const float alpha = __expf(mrow[s][r] - mnew);
          mrow[s][r] = mnew;
          const float p0 = keep0 ? __expf(v0 - mnew) : 0.0f;
          const float p1 = keep1 ? __expf(v1 - mnew) : 0.0f;
          float rsum = p0 + p1;
          rsum += __shfl_xor(rsum, 1, 32);
          rsum += __shfl_xor(rsum, 2, 32);
          rsum += __shfl_xor(rsum, 4, 32);
          rsum += __shfl_xor(rsum, 8, 32);
          lrow[s][r] = lrow[s][r] * alpha + rsum;
#pragma unroll
          for (int t = 0; t < 4; ++t) o[s][t][r] *= alpha;
          const int mloc = s * 16 + hs * 8 + r;
          pb[mloc * 32 + l15]      = f2bf(p0);  // D-layout -> row-major in LDS
          pb[mloc * 32 + 16 + l15] = f2bf(p1);
        }
      }
      asm volatile("s_wait_dscnt 0" ::: "memory");
      // O += P V : P reloaded in A-layout from LDS; V-frags shared by strips
      v16bf ap0 = load_frag(pb,       0, 32, 0, lane);
      v16bf ap1 = load_frag(pb + 512, 0, 32, 0, lane);
#pragma unroll
      for (int t = 0; t < 4; ++t) {
        v16bf bv = load_frag(ldsV, t * 16, 32, 0, lane);
        o[0][t] = wmma_bf16(ap0, bv, o[0][t]);
        o[1][t] = wmma_bf16(ap1, bv, o[1][t]);
      }
    }
    asm volatile("s_wait_dscnt 0" ::: "memory");
    __syncthreads();   // everyone done reading ldsK/ldsV before restaging
  }

#pragma unroll
  for (int s = 0; s < 2; ++s) {
#pragma unroll
    for (int r = 0; r < 8; ++r) {
      const float inv = 1.0f / lrow[s][r];
      const int tq = qs + s * 16 + hs * 8 + r;
      const size_t rowoff = ((size_t)b * T_ + tq) * C_ + h * HD_ + l15;
#pragma unroll
      for (int t = 0; t < 4; ++t) attn[rowoff + t * 16] = f2bf(o[s][t][r] * inv);
    }
  }
}

// ------------------------------------------------------------ out GEMM
// [8192,1024] x [1024,1024] -> fp32 out.  Same 32x64-per-wave blocking.
__global__ __launch_bounds__(256) void k_out_gemm(
    const unsigned short* __restrict__ ab,   // [8192][1024] bf16
    const unsigned short* __restrict__ wt,   // [1024][1024] bf16 (N-major)
    const float* __restrict__ bias,          // [1024]
    float* __restrict__ out) {
  const int lane = threadIdx.x & 31;
  const int wave = threadIdx.x >> 5;
  const int wm = wave >> 2, wn = wave & 3;
  const int rm = blockIdx.y * 64 + wm * 32;
  const int cn = blockIdx.x * 256 + wn * 64;
  v8f acc[2][4];
#pragma unroll
  for (int i = 0; i < 2; ++i)
#pragma unroll
    for (int j = 0; j < 4; ++j) acc[i][j] = v8f_zero();

#pragma unroll 2
  for (int k0 = 0; k0 < C_; k0 += 32) {
    v16bf fa[2], fb[4];
#pragma unroll
    for (int i = 0; i < 2; ++i) fa[i] = load_frag(ab, rm + i * 16, C_, k0, lane);
#pragma unroll
    for (int j = 0; j < 4; ++j) fb[j] = load_frag(wt, cn + j * 16, C_, k0, lane);
    if (k0 + 64 < C_) {
      __builtin_prefetch(ab + (size_t)(rm + (lane & 15)) * C_ + k0 + 64, 0, 1);
      __builtin_prefetch(wt + (size_t)(cn + (lane & 15)) * C_ + k0 + 64, 0, 1);
    }
#pragma unroll
    for (int i = 0; i < 2; ++i)
#pragma unroll
      for (int j = 0; j < 4; ++j) acc[i][j] = wmma_bf16(fa[i], fb[j], acc[i][j]);
  }

  const int l15 = lane & 15, hs = lane >> 4;
#pragma unroll
  for (int i = 0; i < 2; ++i) {
#pragma unroll
    for (int j = 0; j < 4; ++j) {
#pragma unroll
      for (int r = 0; r < 8; ++r) {
        const int m = rm + i * 16 + hs * 8 + r;
        const int n = cn + j * 16 + l15;
        out[(size_t)m * C_ + n] = acc[i][j][r] + bias[n];
      }
    }
  }
}

extern "C" void kernel_launch(void* const* d_in, const int* in_sizes, int n_in,
                              void* d_out, int out_size, void* d_ws, size_t ws_size,
                              hipStream_t stream) {
  (void)in_sizes; (void)n_in; (void)out_size; (void)ws_size;
  const float* x     = (const float*)d_in[0];
  const int*   pad   = (const int*)d_in[1];
  const float* W_qkv = (const float*)d_in[2];
  const float* b_qkv = (const float*)d_in[3];
  const float* W_out = (const float*)d_in[4];
  const float* b_out = (const float*)d_in[5];
  float* out = (float*)d_out;

  char* ws = (char*)d_ws;
  unsigned short* xb   = (unsigned short*)(ws);               // 16 MB
  unsigned short* wqt  = (unsigned short*)(ws + 16777216);    //  6 MB
  unsigned short* wot  = (unsigned short*)(ws + 23068672);    //  2 MB
  unsigned short* qws  = (unsigned short*)(ws + 25165824);    // 16 MB
  unsigned short* kws  = (unsigned short*)(ws + 41943040);    // 16 MB
  unsigned short* vtws = (unsigned short*)(ws + 58720256);    // 16 MB
  unsigned short* attn = (unsigned short*)(ws + 75497472);    // 16 MB

  k_cvt  <<<2048, 256, 0, stream>>>(x, xb, B_ * T_ * C_);
  k_cvt_t<<<2048, 256, 0, stream>>>(W_qkv, wqt, C_, 3 * C_);
  k_cvt_t<<<1024, 256, 0, stream>>>(W_out, wot, C_, C_);
  k_qkv_gemm<<<dim3(12, 128), 256, 0, stream>>>(xb, wqt, b_qkv, qws, kws, vtws);
  k_attn    <<<dim3(T_ / 128, B_ * H_), 128, 0, stream>>>(qws, kws, vtws, pad, attn);
  k_out_gemm<<<dim3(4, 128), 256, 0, stream>>>(attn, wot, b_out, out);
}
